// HiDreamImageSingleTransformerBlock_15298673508874
// MI455X (gfx1250) — compile-verified, matrix operand order checked by reading, not run on hardware
//
#include <hip/hip_runtime.h>
#include <hip/hip_bf16.h>
#include <math.h>

#define DIMC   2560
#define SEQ    2048
#define NHEADS 20
#define HDIM   128
#define NEXP   4
#define HIDC   6912
#define SHIDC  3584
#define CAP    4608          // padded routed-slot capacity (2*SEQ + 4*128)
#define MODLEN (6*DIMC)

typedef __bf16 bf16;
typedef __bf16 v16bf  __attribute__((ext_vector_type(16)));
typedef float  v8f    __attribute__((ext_vector_type(8)));
typedef float  f32x4  __attribute__((ext_vector_type(4)));
typedef float  f32x2  __attribute__((ext_vector_type(2)));

__device__ __forceinline__ v8f wmma_bf16(v16bf a, v16bf b, v8f c) {
  return __builtin_amdgcn_wmma_f32_16x16x32_bf16(false, a, false, b, (short)0, c, false, false);
}

// CDNA5 async global->LDS copy (16B per lane), tracked by ASYNCcnt.
__device__ __forceinline__ void async_ld16(void* lds, const void* gsrc) {
  asm volatile("global_load_async_to_lds_b128 %0, %1, off"
               :: "v"((unsigned)(size_t)lds), "v"(gsrc)
               : "memory");
}
__device__ __forceinline__ void wait_async0() {
  asm volatile("s_wait_asynccnt 0x0" ::: "memory");
}

union Frag16 { v16bf f; f32x4 q[2]; };
union Pack8  { f32x4 q; bf16 b[8]; };
union Pack4  { f32x2 q; bf16 b[4]; };

// Load a 16x32 A-style fragment from a row-major bf16 tile in LDS.
// ISA 7.12.2 (16-bit A 16x32): lane&15 = row; lane>=16 selects K+8 halves;
// components 0..7 = K [kb+kh*8 .. +8), components 8..15 = K [kb+16+kh*8 .. +8).
__device__ __forceinline__ v16bf ld_frag(const bf16* tile, int ld, int rowbase, int kb, int lane) {
  Frag16 u;
  const bf16* p = tile + (size_t)(rowbase + (lane & 15)) * ld + kb + ((lane >> 4) << 3);
  u.q[0] = *(const f32x4*)p;
  u.q[1] = *(const f32x4*)(p + 16);
  return u.f;
}

__device__ __forceinline__ float waveReduceSum(float v) {
#pragma unroll
  for (int off = 16; off; off >>= 1) v += __shfl_xor(v, off, 32);
  return v;
}
__device__ __forceinline__ float halfReduceMax(float v) {
  v = fmaxf(v, __shfl_xor(v, 1, 32));
  v = fmaxf(v, __shfl_xor(v, 2, 32));
  v = fmaxf(v, __shfl_xor(v, 4, 32));
  v = fmaxf(v, __shfl_xor(v, 8, 32));
  return v;
}
__device__ __forceinline__ float halfReduceSum(float v) {
  v += __shfl_xor(v, 1, 32);
  v += __shfl_xor(v, 2, 32);
  v += __shfl_xor(v, 4, 32);
  v += __shfl_xor(v, 8, 32);
  return v;
}

// ---------------- small elementwise / GEMV kernels ----------------

__global__ __launch_bounds__(256) void silu_vec(const float* __restrict__ x, float* __restrict__ y) {
  int i = blockIdx.x * 256 + threadIdx.x;
  if (i < DIMC) { float v = x[i]; y[i] = v / (1.f + __expf(-v)); }
}

__global__ __launch_bounds__(256) void adaln_gemv(const float* __restrict__ st,
                                                  const float* __restrict__ W,
                                                  const float* __restrict__ b,
                                                  float* __restrict__ mod) {
  int lane = threadIdx.x & 31;
  int row  = blockIdx.x * 8 + (threadIdx.x >> 5);
  const float* wr = W + (size_t)row * DIMC;
  float s = 0.f;
  for (int c = lane; c < DIMC; c += 32) s += wr[c] * st[c];
  s = waveReduceSum(s);
  if (lane == 0) mod[row] = s + b[row];
}

__global__ __launch_bounds__(256) void ln_mod(const float* __restrict__ X,
                                              const float* __restrict__ shift,
                                              const float* __restrict__ scale,
                                              bf16* __restrict__ Yb,
                                              float* __restrict__ Yf) {
  __shared__ float sm[16];
  int t = blockIdx.x;
  const float* x = X + (size_t)t * DIMC;
  float s1 = 0.f, s2 = 0.f;
  for (int c = threadIdx.x; c < DIMC; c += 256) { float a = x[c]; s1 += a; s2 += a * a; }
  s1 = waveReduceSum(s1); s2 = waveReduceSum(s2);
  int wid = threadIdx.x >> 5, lane = threadIdx.x & 31;
  if (lane == 0) { sm[wid] = s1; sm[8 + wid] = s2; }
  __syncthreads();
  float t1 = 0.f, t2 = 0.f;
#pragma unroll
  for (int i = 0; i < 8; ++i) { t1 += sm[i]; t2 += sm[8 + i]; }
  float m = t1 / DIMC;
  float var = t2 / DIMC - m * m;
  float r = rsqrtf(var + 1e-6f);
  for (int c = threadIdx.x; c < DIMC; c += 256) {
    float y = (x[c] - m) * r * (1.f + scale[c]) + shift[c];
    Yb[(size_t)t * DIMC + c] = (bf16)y;
    if (Yf) Yf[(size_t)t * DIMC + c] = y;
  }
}

// RMSNorm(q,k over full 2560) + RoPE + pack per-head bf16 Q(pre-scaled)/K/V
__global__ __launch_bounds__(256) void rms_rope(const float* __restrict__ qkv,
                                                const float* __restrict__ rope,
                                                const float* __restrict__ qn,
                                                const float* __restrict__ kn,
                                                bf16* __restrict__ Qb, bf16* __restrict__ Kb,
                                                bf16* __restrict__ Vb) {
  __shared__ float sm[16];
  int t = blockIdx.x;
  const float* row = qkv + (size_t)t * 3 * DIMC;
  float sq = 0.f, sk = 0.f;
  for (int c = threadIdx.x; c < DIMC; c += 256) {
    float a = row[c];        sq += a * a;
    float b = row[DIMC + c]; sk += b * b;
  }
  sq = waveReduceSum(sq); sk = waveReduceSum(sk);
  int wid = threadIdx.x >> 5, lane = threadIdx.x & 31;
  if (lane == 0) { sm[wid] = sq; sm[8 + wid] = sk; }
  __syncthreads();
  float t1 = 0.f, t2 = 0.f;
#pragma unroll
  for (int i = 0; i < 8; ++i) { t1 += sm[i]; t2 += sm[8 + i]; }
  float rq = rsqrtf(t1 / DIMC + 1e-6f);
  float rk = rsqrtf(t2 / DIMC + 1e-6f);
  const float rs = rsqrtf((float)HDIM);     // softmax scale folded into Q
  for (int p = threadIdx.x; p < DIMC / 2; p += 256) {
    int hh = p / 64, d2 = p % 64, c = 2 * p;
    const float* rp = rope + (((size_t)t * 64 + d2) << 2);
    float r00 = rp[0], r01 = rp[1], r10 = rp[2], r11 = rp[3];
    float q0 = row[c] * rq * qn[c],        q1 = row[c + 1] * rq * qn[c + 1];
    float k0 = row[DIMC + c] * rk * kn[c], k1 = row[DIMC + c + 1] * rk * kn[c + 1];
    size_t base = ((size_t)hh * SEQ + t) * HDIM + 2 * d2;
    Qb[base]     = (bf16)((r00 * q0 + r01 * q1) * rs);
    Qb[base + 1] = (bf16)((r10 * q0 + r11 * q1) * rs);
    Kb[base]     = (bf16)(r00 * k0 + r01 * k1);
    Kb[base + 1] = (bf16)(r10 * k0 + r11 * k1);
    Vb[base]     = (bf16)row[2 * DIMC + c];
    Vb[base + 1] = (bf16)row[2 * DIMC + c + 1];
  }
}

// ---------------- WMMA GEMM: C[M,N] = A[M,K](bf16) * B[N,K]^T(f32->bf16) ----------------
// Async double-buffered A tiles (global->LDS DMA) + software-pipelined B conversion.
// MODE 0: +bias (QKV). MODE 1: gate*(C+bias)+resid (out-proj+residual). MODE 2: plain.
template <int MODE>
__global__ __launch_bounds__(256) void gemm_nt(const bf16* __restrict__ A, int lda,
                                               const float* __restrict__ B, size_t bstride,
                                               float* __restrict__ C, int ldc, int K,
                                               const float* __restrict__ bias,
                                               const float* __restrict__ gate,
                                               const float* __restrict__ resid,
                                               const int* __restrict__ counts,
                                               const int* __restrict__ seg) {
  __shared__ bf16 As[2][128][72];
  __shared__ bf16 Bs[2][64][72];
  const int z  = blockIdx.z;
  const int m0 = blockIdx.y * 128;
  const int n0 = blockIdx.x * 64;
  if (counts && m0 >= counts[z]) return;
  const int rowoff = seg ? seg[z] : 0;
  const float* Bp = B + (size_t)z * bstride;
  const int tid = threadIdx.x, lane = tid & 31, wid = tid >> 5;
  const int wm = wid & 3, wn = wid >> 2;

  int ar[4], ac[4], br[4], bc[4];
#pragma unroll
  for (int i = 0; i < 4; ++i) {
    int idx = tid + i * 256;
    ar[i] = idx >> 3;  ac[i] = (idx & 7) << 3;
    br[i] = idx >> 4;  bc[i] = (idx & 15) << 2;
  }
  auto issueA = [&](int kt, int buf) {
    int k0 = kt << 6;
#pragma unroll
    for (int i = 0; i < 4; ++i)
      async_ld16(&As[buf][ar[i]][ac[i]],
                 A + (size_t)(rowoff + m0 + ar[i]) * lda + k0 + ac[i]);
  };
  auto loadB = [&](int kt, f32x4* wv) {
    int k0 = kt << 6;
#pragma unroll
    for (int i = 0; i < 4; ++i)
      wv[i] = *(const f32x4*)(Bp + (size_t)(n0 + br[i]) * K + k0 + bc[i]);
  };
  auto storeB = [&](const f32x4* wv, int buf) {
#pragma unroll
    for (int i = 0; i < 4; ++i) {
      Pack4 pk;
      pk.b[0] = (bf16)wv[i].x; pk.b[1] = (bf16)wv[i].y;
      pk.b[2] = (bf16)wv[i].z; pk.b[3] = (bf16)wv[i].w;
      *(f32x2*)&Bs[buf][br[i]][bc[i]] = pk.q;
    }
  };

  v8f acc[2][2];
#pragma unroll
  for (int i = 0; i < 2; ++i)
#pragma unroll
    for (int j = 0; j < 2; ++j)
#pragma unroll
      for (int v = 0; v < 8; ++v) acc[i][j][v] = 0.f;

  const int nkt = K >> 6;
  f32x4 wv[4];
  issueA(0, 0);
  loadB(0, wv);
  storeB(wv, 0);
  wait_async0();
  __syncthreads();

  for (int kt = 0; kt < nkt; ++kt) {
    const int cur = kt & 1, nxt = cur ^ 1;
    const bool more = (kt + 1) < nkt;
    if (more) {
      issueA(kt + 1, nxt);               // DMA next A tile while computing
      loadB(kt + 1, wv);                 // f32 weight loads overlap WMMA
      if (kt + 2 < nkt)
        __builtin_prefetch(Bp + (size_t)(n0 + br[0]) * K + ((kt + 2) << 6) + bc[0], 0, 1);
    }
#pragma unroll
    for (int ks = 0; ks < 2; ++ks) {
      int kb = ks << 5;
      v16bf a0 = ld_frag(&As[cur][0][0], 72, wm * 32,      kb, lane);
      v16bf a1 = ld_frag(&As[cur][0][0], 72, wm * 32 + 16, kb, lane);
      v16bf b0 = ld_frag(&Bs[cur][0][0], 72, wn * 32,      kb, lane);
      v16bf b1 = ld_frag(&Bs[cur][0][0], 72, wn * 32 + 16, kb, lane);
      acc[0][0] = wmma_bf16(a0, b0, acc[0][0]);
      acc[0][1] = wmma_bf16(a0, b1, acc[0][1]);
      acc[1][0] = wmma_bf16(a1, b0, acc[1][0]);
      acc[1][1] = wmma_bf16(a1, b1, acc[1][1]);
    }
    if (more) storeB(wv, nxt);
    wait_async0();
    __syncthreads();
  }

  const int m8 = (lane >> 4) << 3, nc = lane & 15;
#pragma unroll
  for (int mi = 0; mi < 2; ++mi)
#pragma unroll
    for (int ni = 0; ni < 2; ++ni) {
      int col = n0 + wn * 32 + ni * 16 + nc;
#pragma unroll
      for (int v = 0; v < 8; ++v) {
        int row = rowoff + m0 + wm * 32 + mi * 16 + m8 + v;
        float val = acc[mi][ni][v];
        if (MODE == 0) {
          C[(size_t)row * ldc + col] = val + bias[col];
        } else if (MODE == 1) {
          C[(size_t)row * ldc + col] =
              gate[col] * (val + bias[col]) + resid[(size_t)row * ldc + col];
        } else {
          C[(size_t)row * ldc + col] = val;
        }
      }
    }
}

// Fused dual up-proj: H[row,n] = bf16( silu(A*W1^T) * (A*W3^T) ), optional row gather.
__global__ __launch_bounds__(256) void dual_up(const bf16* __restrict__ A, int lda,
                                               const float* __restrict__ W1,
                                               const float* __restrict__ W3, size_t bstride,
                                               bf16* __restrict__ H, int ldh, int K,
                                               const int* __restrict__ rtok,
                                               const int* __restrict__ counts,
                                               const int* __restrict__ seg) {
  __shared__ bf16 As[2][128][72];
  __shared__ bf16 B1s[2][64][72];
  __shared__ bf16 B3s[2][64][72];
  const int z  = blockIdx.z;
  const int m0 = blockIdx.y * 128;
  const int n0 = blockIdx.x * 64;
  if (counts && m0 >= counts[z]) return;
  const int rowoff = seg ? seg[z] : 0;
  const float* W1p = W1 + (size_t)z * bstride;
  const float* W3p = W3 + (size_t)z * bstride;
  const int tid = threadIdx.x, lane = tid & 31, wid = tid >> 5;
  const int wm = wid & 3, wn = wid >> 2;

  int ar[4], ac[4], br[4], bc[4], grows[4];
#pragma unroll
  for (int i = 0; i < 4; ++i) {
    int idx = tid + i * 256;
    ar[i] = idx >> 3;  ac[i] = (idx & 7) << 3;
    br[i] = idx >> 4;  bc[i] = (idx & 15) << 2;
    grows[i] = rtok ? rtok[rowoff + m0 + ar[i]] : (m0 + ar[i]);
  }
  auto issueA = [&](int kt, int buf) {
    int k0 = kt << 6;
#pragma unroll
    for (int i = 0; i < 4; ++i)
      async_ld16(&As[buf][ar[i]][ac[i]], A + (size_t)grows[i] * lda + k0 + ac[i]);
  };
  auto loadB = [&](int kt, f32x4* w1v, f32x4* w3v) {
    int k0 = kt << 6;
#pragma unroll
    for (int i = 0; i < 4; ++i) {
      w1v[i] = *(const f32x4*)(W1p + (size_t)(n0 + br[i]) * K + k0 + bc[i]);
      w3v[i] = *(const f32x4*)(W3p + (size_t)(n0 + br[i]) * K + k0 + bc[i]);
    }
  };
  auto storeB = [&](const f32x4* w1v, const f32x4* w3v, int buf) {
#pragma unroll
    for (int i = 0; i < 4; ++i) {
      Pack4 p1, p3;
      p1.b[0] = (bf16)w1v[i].x; p1.b[1] = (bf16)w1v[i].y;
      p1.b[2] = (bf16)w1v[i].z; p1.b[3] = (bf16)w1v[i].w;
      p3.b[0] = (bf16)w3v[i].x; p3.b[1] = (bf16)w3v[i].y;
      p3.b[2] = (bf16)w3v[i].z; p3.b[3] = (bf16)w3v[i].w;
      *(f32x2*)&B1s[buf][br[i]][bc[i]] = p1.q;
      *(f32x2*)&B3s[buf][br[i]][bc[i]] = p3.q;
    }
  };

  v8f acc1[2][2], acc3[2][2];
#pragma unroll
  for (int i = 0; i < 2; ++i)
#pragma unroll
    for (int j = 0; j < 2; ++j)
#pragma unroll
      for (int v = 0; v < 8; ++v) { acc1[i][j][v] = 0.f; acc3[i][j][v] = 0.f; }

  const int nkt = K >> 6;
  f32x4 w1v[4], w3v[4];
  issueA(0, 0);
  loadB(0, w1v, w3v);
  storeB(w1v, w3v, 0);
  wait_async0();
  __syncthreads();

  for (int kt = 0; kt < nkt; ++kt) {
    const int cur = kt & 1, nxt = cur ^ 1;
    const bool more = (kt + 1) < nkt;
    if (more) {
      issueA(kt + 1, nxt);
      loadB(kt + 1, w1v, w3v);
      if (kt + 2 < nkt) {
        __builtin_prefetch(W1p + (size_t)(n0 + br[0]) * K + ((kt + 2) << 6) + bc[0], 0, 1);
        __builtin_prefetch(W3p + (size_t)(n0 + br[0]) * K + ((kt + 2) << 6) + bc[0], 0, 1);
      }
    }
#pragma unroll
    for (int ks = 0; ks < 2; ++ks) {
      int kb = ks << 5;
      v16bf a0 = ld_frag(&As[cur][0][0], 72, wm * 32,      kb, lane);
      v16bf a1 = ld_frag(&As[cur][0][0], 72, wm * 32 + 16, kb, lane);
      v16bf c0 = ld_frag(&B1s[cur][0][0], 72, wn * 32,      kb, lane);
      v16bf c1 = ld_frag(&B1s[cur][0][0], 72, wn * 32 + 16, kb, lane);
      v16bf d0 = ld_frag(&B3s[cur][0][0], 72, wn * 32,      kb, lane);
      v16bf d1 = ld_frag(&B3s[cur][0][0], 72, wn * 32 + 16, kb, lane);
      acc1[0][0] = wmma_bf16(a0, c0, acc1[0][0]);
      acc1[0][1] = wmma_bf16(a0, c1, acc1[0][1]);
      acc1[1][0] = wmma_bf16(a1, c0, acc1[1][0]);
      acc1[1][1] = wmma_bf16(a1, c1, acc1[1][1]);
      acc3[0][0] = wmma_bf16(a0, d0, acc3[0][0]);
      acc3[0][1] = wmma_bf16(a0, d1, acc3[0][1]);
      acc3[1][0] = wmma_bf16(a1, d0, acc3[1][0]);
      acc3[1][1] = wmma_bf16(a1, d1, acc3[1][1]);
    }
    if (more) storeB(w1v, w3v, nxt);
    wait_async0();
    __syncthreads();
  }

  const int m8 = (lane >> 4) << 3, nc = lane & 15;
#pragma unroll
  for (int mi = 0; mi < 2; ++mi)
#pragma unroll
    for (int ni = 0; ni < 2; ++ni) {
      int col = n0 + wn * 32 + ni * 16 + nc;
#pragma unroll
      for (int v = 0; v < 8; ++v) {
        int row = rowoff + m0 + wm * 32 + mi * 16 + m8 + v;
        float g = acc1[mi][ni][v], u3 = acc3[mi][ni][v];
        float sg = g / (1.f + __expf(-g));
        H[(size_t)row * ldh + col] = (bf16)(sg * u3);
      }
    }
}

// ---------------- Flash attention (per head, 64 queries/block, 4 waves x 16 rows) ----------------
__global__ __launch_bounds__(128) void attn_fa(const bf16* __restrict__ Q,
                                               const bf16* __restrict__ Km,
                                               const bf16* __restrict__ Vm,
                                               bf16* __restrict__ O) {
  __shared__ bf16 Qs[64][HDIM + 8];
  __shared__ bf16 Ks[32][HDIM + 8];
  __shared__ bf16 Vts[HDIM][40];
  __shared__ bf16 Ps[4][16][40];
  const int h = blockIdx.y, qt = blockIdx.x;
  const int tid = threadIdx.x, lane = tid & 31, w = tid >> 5;
  const bf16* Qh = Q + ((size_t)h * SEQ + qt * 64) * HDIM;
  const bf16* Kh = Km + (size_t)h * SEQ * HDIM;
  const bf16* Vh = Vm + (size_t)h * SEQ * HDIM;

  for (int i = tid; i < 64 * HDIM / 8; i += 128) {        // async Q tile DMA
    int r = i >> 4, c = (i & 15) << 3;
    async_ld16(&Qs[r][c], Qh + (size_t)r * HDIM + c);
  }
  v8f o[8];
  float l[8], mo[8];
#pragma unroll
  for (int f = 0; f < 8; ++f)
#pragma unroll
    for (int v = 0; v < 8; ++v) o[f][v] = 0.f;
#pragma unroll
  for (int v = 0; v < 8; ++v) { l[v] = 0.f; mo[v] = -1e30f; }

  for (int kt = 0; kt < SEQ / 32; ++kt) {
    __syncthreads();
    const bf16* Kt = Kh + (size_t)kt * 32 * HDIM;
    const bf16* Vt = Vh + (size_t)kt * 32 * HDIM;
    for (int i = tid; i < 32 * HDIM / 8; i += 128) {
      int r = i >> 4, c = (i & 15) << 3;
      async_ld16(&Ks[r][c], Kt + (size_t)r * HDIM + c);   // async K tile DMA
      Pack8 u;
      u.q = *(const f32x4*)(Vt + (size_t)r * HDIM + c);
#pragma unroll
      for (int j = 0; j < 8; ++j) Vts[c + j][r] = u.b[j]; // transpose V -> [d][key]
    }
    wait_async0();
    __syncthreads();

    v8f s0, s1;
#pragma unroll
    for (int v = 0; v < 8; ++v) { s0[v] = 0.f; s1[v] = 0.f; }
#pragma unroll
    for (int kb = 0; kb < HDIM; kb += 32) {
      v16bf aq  = ld_frag(&Qs[0][0], HDIM + 8, w * 16, kb, lane);
      v16bf bk0 = ld_frag(&Ks[0][0], HDIM + 8, 0,      kb, lane);
      v16bf bk1 = ld_frag(&Ks[0][0], HDIM + 8, 16,     kb, lane);
      s0 = wmma_bf16(aq, bk0, s0);
      s1 = wmma_bf16(aq, bk1, s1);
    }
    const int hi = lane >> 4;
#pragma unroll
    for (int v = 0; v < 8; ++v) {
      float tmax = halfReduceMax(fmaxf(s0[v], s1[v]));
      float mn = fmaxf(mo[v], tmax);
      float al = __expf(mo[v] - mn);
      mo[v] = mn;
      float p0 = __expf(s0[v] - mn);
      float p1 = __expf(s1[v] - mn);
      float rs = halfReduceSum(p0 + p1);
      l[v] = l[v] * al + rs;
#pragma unroll
      for (int f = 0; f < 8; ++f) o[f][v] *= al;
      int m = v + (hi << 3);
      Ps[w][m][lane & 15]        = (bf16)p0;
      Ps[w][m][16 + (lane & 15)] = (bf16)p1;
    }
    // P(16x32) * V(32x128): DS ops are in-order within the wave
    v16bf ap = ld_frag(&Ps[w][0][0], 40, 0, 0, lane);
#pragma unroll
    for (int f = 0; f < 8; ++f) {
      v16bf bv = ld_frag(&Vts[0][0], 40, f * 16, 0, lane);
      o[f] = wmma_bf16(ap, bv, o[f]);
    }
  }
  const int nc = lane & 15, hi = lane >> 4;
#pragma unroll
  for (int v = 0; v < 8; ++v) {
    float inv = 1.f / l[v];
    int t = qt * 64 + w * 16 + v + (hi << 3);
#pragma unroll
    for (int f = 0; f < 8; ++f)
      O[(size_t)t * DIMC + h * HDIM + f * 16 + nc] = (bf16)(o[f][v] * inv);
  }
}

// ---------------- MoE routing ----------------

__global__ __launch_bounds__(256) void route_init(int* counts, int* cursor, int* rtok, float* rw) {
  int i = blockIdx.x * 256 + threadIdx.x;
  if (i < CAP) { rtok[i] = 0; rw[i] = 0.f; }
  if (i < NEXP) { counts[i] = 0; cursor[i] = 0; }
}

__global__ __launch_bounds__(128) void gate_topk(const float* __restrict__ X,
                                                 const float* __restrict__ gw,
                                                 float* __restrict__ tw, int* __restrict__ ti,
                                                 int* __restrict__ counts) {
  __shared__ float sc[NEXP];
  int t = blockIdx.x, w = threadIdx.x >> 5, lane = threadIdx.x & 31;
  const float* x = X + (size_t)t * DIMC;
  const float* g = gw + (size_t)w * DIMC;
  float s = 0.f;
  for (int c = lane; c < DIMC; c += 32) s += x[c] * g[c];
  s = waveReduceSum(s);
  if (lane == 0) sc[w] = s;
  __syncthreads();
  if (threadIdx.x == 0) {
    float m = fmaxf(fmaxf(sc[0], sc[1]), fmaxf(sc[2], sc[3]));
    float e[NEXP], d = 0.f;
#pragma unroll
    for (int i = 0; i < NEXP; ++i) { e[i] = __expf(sc[i] - m); d += e[i]; }
    float p[NEXP];
#pragma unroll
    for (int i = 0; i < NEXP; ++i) p[i] = e[i] / d;
    int i0 = 0;
#pragma unroll
    for (int i = 1; i < NEXP; ++i) if (p[i] > p[i0]) i0 = i;
    int i1 = -1;
#pragma unroll
    for (int i = 0; i < NEXP; ++i) if (i != i0 && (i1 < 0 || p[i] > p[i1])) i1 = i;
    ti[2 * t] = i0; ti[2 * t + 1] = i1;
    tw[2 * t] = p[i0]; tw[2 * t + 1] = p[i1];
    atomicAdd(counts + i0, 1);
    atomicAdd(counts + i1, 1);
  }
}

__global__ void seg_scan(const int* counts, int* seg) {
  if (threadIdx.x == 0) {
    int o = 0;
    for (int e = 0; e < NEXP; ++e) { seg[e] = o; o += ((counts[e] + 127) >> 7) << 7; }
  }
}

__global__ __launch_bounds__(256) void assign_k(const int* __restrict__ ti,
                                                const float* __restrict__ tw,
                                                const int* __restrict__ seg, int* cursor,
                                                int* __restrict__ rtok, float* __restrict__ rw,
                                                int* __restrict__ slots) {
  int t = blockIdx.x * 256 + threadIdx.x;
  if (t >= SEQ) return;
#pragma unroll
  for (int j = 0; j < 2; ++j) {
    int e = ti[2 * t + j];
    int pos = atomicAdd(cursor + e, 1);
    int s = seg[e] + pos;
    rtok[s] = t;
    rw[s] = tw[2 * t + j];
    slots[2 * t + j] = s;
  }
}

__global__ __launch_bounds__(256) void combine(const float* __restrict__ outr,
                                               const float* __restrict__ sho,
                                               const float* __restrict__ rw,
                                               const int* __restrict__ slots,
                                               const float* __restrict__ gmlp,
                                               const float* __restrict__ h2,
                                               float* __restrict__ out) {
  int t = blockIdx.x;
  int s0 = slots[2 * t], s1 = slots[2 * t + 1];
  float w0 = rw[s0], w1 = rw[s1];
  for (int c = threadIdx.x; c < DIMC; c += 256) {
    float y = w0 * outr[(size_t)s0 * DIMC + c] + w1 * outr[(size_t)s1 * DIMC + c] +
              sho[(size_t)t * DIMC + c];
    out[(size_t)t * DIMC + c] = gmlp[c] * y + h2[(size_t)t * DIMC + c];
  }
}

// ---------------- host launch ----------------

extern "C" void kernel_launch(void* const* d_in, const int* in_sizes, int n_in,
                              void* d_out, int out_size, void* d_ws, size_t ws_size,
                              hipStream_t stream) {
  (void)in_sizes; (void)n_in; (void)out_size; (void)ws_size;
  const float* hidden = (const float*)d_in[0];
  const float* temb   = (const float*)d_in[1];
  const float* rope   = (const float*)d_in[2];
  const float* adaw   = (const float*)d_in[3];
  const float* adab   = (const float*)d_in[4];
  const float* qkvw   = (const float*)d_in[5];
  const float* qkvb   = (const float*)d_in[6];
  const float* qnw    = (const float*)d_in[7];
  const float* knw    = (const float*)d_in[8];
  const float* outw   = (const float*)d_in[9];
  const float* outb   = (const float*)d_in[10];
  const float* gatew  = (const float*)d_in[11];
  const float* ew1    = (const float*)d_in[12];
  const float* ew2    = (const float*)d_in[13];
  const float* ew3    = (const float*)d_in[14];
  const float* sw1    = (const float*)d_in[15];
  const float* sw2    = (const float*)d_in[16];
  const float* sw3    = (const float*)d_in[17];
  float* out = (float*)d_out;

  char* w = (char*)d_ws;
  size_t off = 0;
  auto alloc = [&](size_t bytes) -> char* {
    char* p = w + off;
    off = (off + bytes + 255) & ~(size_t)255;
    return p;
  };
  float* mod   = (float*)alloc((size_t)MODLEN * 4);
  float* st    = (float*)alloc((size_t)DIMC * 4);
  bf16*  nh1   = (bf16*) alloc((size_t)SEQ * DIMC * 2);
  float* qkv   = (float*)alloc((size_t)SEQ * 3 * DIMC * 4);
  bf16*  qbf   = (bf16*) alloc((size_t)NHEADS * SEQ * HDIM * 2);
  bf16*  kbf   = (bf16*) alloc((size_t)NHEADS * SEQ * HDIM * 2);
  bf16*  vbf   = (bf16*) alloc((size_t)NHEADS * SEQ * HDIM * 2);
  bf16*  ob    = (bf16*) alloc((size_t)SEQ * DIMC * 2);
  float* h2    = (float*)alloc((size_t)SEQ * DIMC * 4);
  bf16*  nh2   = (bf16*) alloc((size_t)SEQ * DIMC * 2);
  float* nh2f  = (float*)alloc((size_t)SEQ * DIMC * 4);
  int*   counts= (int*)  alloc(NEXP * 4);
  int*   cursor= (int*)  alloc(NEXP * 4);
  int*   seg   = (int*)  alloc(NEXP * 4);
  int*   ti    = (int*)  alloc((size_t)SEQ * 2 * 4);
  float* tw    = (float*)alloc((size_t)SEQ * 2 * 4);
  int*   slots = (int*)  alloc((size_t)SEQ * 2 * 4);
  int*   rtok  = (int*)  alloc((size_t)CAP * 4);
  float* rw    = (float*)alloc((size_t)CAP * 4);
  bf16*  hb    = (bf16*) alloc((size_t)CAP * HIDC * 2);
  float* outr  = (float*)alloc((size_t)CAP * DIMC * 4);
  bf16*  hs    = (bf16*) alloc((size_t)SEQ * SHIDC * 2);
  float* sho   = (float*)alloc((size_t)SEQ * DIMC * 4);

  // adaLN modulation
  silu_vec<<<(DIMC + 255) / 256, 256, 0, stream>>>(temb, st);
  adaln_gemv<<<MODLEN / 8, 256, 0, stream>>>(st, adaw, adab, mod);
  // attention branch
  ln_mod<<<SEQ, 256, 0, stream>>>(hidden, mod, mod + DIMC, nh1, (float*)nullptr);
  gemm_nt<0><<<dim3(3 * DIMC / 64, SEQ / 128, 1), 256, 0, stream>>>(
      nh1, DIMC, qkvw, 0, qkv, 3 * DIMC, DIMC, qkvb, nullptr, nullptr, nullptr, nullptr);
  rms_rope<<<SEQ, 256, 0, stream>>>(qkv, rope, qnw, knw, qbf, kbf, vbf);
  attn_fa<<<dim3(SEQ / 64, NHEADS), 128, 0, stream>>>(qbf, kbf, vbf, ob);
  gemm_nt<1><<<dim3(DIMC / 64, SEQ / 128, 1), 256, 0, stream>>>(
      ob, DIMC, outw, 0, h2, DIMC, DIMC, outb, mod + 2 * DIMC, hidden, nullptr, nullptr);
  // MoE branch
  ln_mod<<<SEQ, 256, 0, stream>>>(h2, mod + 3 * DIMC, mod + 4 * DIMC, nh2, nh2f);
  route_init<<<(CAP + 255) / 256, 256, 0, stream>>>(counts, cursor, rtok, rw);
  gate_topk<<<SEQ, 128, 0, stream>>>(nh2f, gatew, tw, ti, counts);
  seg_scan<<<1, 1, 0, stream>>>(counts, seg);
  assign_k<<<(SEQ + 255) / 256, 256, 0, stream>>>(ti, tw, seg, cursor, rtok, rw, slots);
  dual_up<<<dim3(HIDC / 64, SEQ / 128, NEXP), 256, 0, stream>>>(
      nh2, DIMC, ew1, ew3, (size_t)HIDC * DIMC, hb, HIDC, DIMC, rtok, counts, seg);
  gemm_nt<2><<<dim3(DIMC / 64, SEQ / 128, NEXP), 256, 0, stream>>>(
      hb, HIDC, ew2, (size_t)DIMC * HIDC, outr, DIMC, HIDC, nullptr, nullptr, nullptr, counts, seg);
  dual_up<<<dim3(SHIDC / 64, SEQ / 128, 1), 256, 0, stream>>>(
      nh2, DIMC, sw1, sw3, 0, hs, SHIDC, DIMC, nullptr, nullptr, nullptr);
  gemm_nt<2><<<dim3(DIMC / 64, SEQ / 128, 1), 256, 0, stream>>>(
      hs, SHIDC, sw2, 0, sho, DIMC, SHIDC, nullptr, nullptr, nullptr, nullptr, nullptr);
  combine<<<SEQ, 256, 0, stream>>>(outr, sho, rw, slots, mod + 5 * DIMC, h2, out);
}
